// SelfAttention_13125420056845
// MI455X (gfx1250) — matrix operand order repaired and checked by reading, not compiled, not run on hardware
//
#include <hip/hip_runtime.h>

// Problem constants (from reference setup_inputs)
#define B_  32
#define S_  512
#define D_  512
#define H_  8
#define DK_ 64
#define NEG_INF_F (-1.0e30f)

typedef _Float16 v16h __attribute__((ext_vector_type(16)));
typedef _Float16 v8h  __attribute__((ext_vector_type(8)));
typedef float    v8f  __attribute__((ext_vector_type(8)));

// ---------------------------------------------------------------------------
// WMMA + async helpers (gfx1250)
// A 16x32 f16 fragment: lane m = lane%16 is the A row; lanes 0-15 hold
//   K = k0..k0+15 contiguous, lanes 16-31 hold K = k0+16..k0+31.
// B 32x16 f16 fragment: lane n = lane%16 is the B column; lanes 0-15 hold
//   K = k0..k0+15 contiguous, lanes 16-31 hold K = k0+16..k0+31.
// C/D f32 16x16: lane%16 = N, vgpr v -> M = v + 8*(lane/16).
// ---------------------------------------------------------------------------
__device__ __forceinline__ v16h load_a16(const _Float16* rowp, int k0, int kbA) {
  v8h lo = *(const v8h*)(rowp + k0 + 2 * kbA);
  v8h hi = *(const v8h*)(rowp + k0 + 2 * kbA + 8);
  v16h a;
#pragma unroll
  for (int i = 0; i < 8; ++i) { a[i] = lo[i]; a[i + 8] = hi[i]; }
  return a;
}

__device__ __forceinline__ v8f wmma16(v16h a, v16h b, v8f c) {
  return __builtin_amdgcn_wmma_f32_16x16x32_f16(false, a, false, b,
                                                (short)0, c, false, false);
}

// Async copy of one 16-byte segment global -> LDS (per-lane), ASYNCcnt-tracked.
// Generic pointers to __shared__ carry the LDS byte address in their low 32
// bits (LDS aperture lives in addr[63:32]), so truncation yields the DS addr.
__device__ __forceinline__ void async_copy16(const void* gptr, void* lptr) {
  unsigned lo = (unsigned)(unsigned long long)lptr;
  asm volatile("global_load_async_to_lds_b128 %0, %1, off"
               :: "v"(lo), "v"(gptr) : "memory");
}
__device__ __forceinline__ void wait_async0() {
  asm volatile("s_wait_asynccnt 0x0" ::: "memory");
}

// ---------------------------------------------------------------------------
// Kernel 1: fp32 -> f16 conversions + weight transposes.
//   xh   [B][S][D]          (row-major, K-contiguous for A fragments)
//   WqT/WkT/WvT [H][DK][D]  (transposed so B fragments are contiguous)
//   WoT  [D][D]             (WoT[n][k] = Wo[k][n])
// ---------------------------------------------------------------------------
__global__ void k_convert(const float* __restrict__ x,
                          const float* __restrict__ Wq,
                          const float* __restrict__ Wk,
                          const float* __restrict__ Wv,
                          const float* __restrict__ Wo,
                          _Float16* __restrict__ xh,
                          _Float16* __restrict__ WqT,
                          _Float16* __restrict__ WkT,
                          _Float16* __restrict__ WvT,
                          _Float16* __restrict__ WoT) {
  const int stride = gridDim.x * blockDim.x;
  const int tid    = blockIdx.x * blockDim.x + threadIdx.x;

  const int NX = B_ * S_ * D_;          // 8388608
  for (int i = tid; i < NX; i += stride) xh[i] = (_Float16)x[i];

  const int NW = H_ * DK_ * D_;         // 262144
  for (int i = tid; i < NW; i += stride) {
    int h = i / (DK_ * D_);
    int r = i - h * (DK_ * D_);
    int n = r / D_;
    int k = r - n * D_;
    int src = (h * D_ + k) * DK_ + n;   // W[h][k][n]
    WqT[i] = (_Float16)Wq[src];
    WkT[i] = (_Float16)Wk[src];
    WvT[i] = (_Float16)Wv[src];
  }

  const int NO = D_ * D_;               // 262144
  for (int i = tid; i < NO; i += stride) {
    int n = i / D_;
    int k = i - n * D_;
    WoT[i] = (_Float16)Wo[k * D_ + n];  // Wo[k][n]
  }
}

// ---------------------------------------------------------------------------
// Kernel 2: fused QKV projection with double-buffered async weight staging.
// All 8 waves of a block share (b,h); the Wq/Wk/Wv k-slice (64 cols x 32 k)
// is async-staged to LDS once per block per k-chunk, double-buffered so the
// copy of chunk k+1 overlaps the 12 WMMAs of chunk k.
// LDS rows padded to 40 halfs (80B = 20 dwords): lane-n fragment reads start
// at bank 20n mod 64 -> all 16 lanes hit disjoint bank quads.
// grid = B*H*(S/16)/8 = 1024 blocks x 256 threads.
// ---------------------------------------------------------------------------
#define WPAD 40
__global__ void k_qkv(const _Float16* __restrict__ xh,
                      const _Float16* __restrict__ WqT,
                      const _Float16* __restrict__ WkT,
                      const _Float16* __restrict__ WvT,
                      const float* __restrict__ bias,
                      _Float16* __restrict__ Qh,   // [B][H][S][DK]
                      _Float16* __restrict__ Kh,   // [B][H][S][DK]
                      _Float16* __restrict__ VT) { // [B][H][DK][S]
  __shared__ __align__(32) _Float16 wbuf[2][3][64 * WPAD];   // 30 KB

  const int tid  = threadIdx.x;
  const int lane = tid & 31;
  const int warp = tid >> 5;
  const int mrow = lane & 15;
  const int hi   = lane >> 4;
  const int kbA  = hi * 8;
  const int kbB  = hi * 16;

  const int bh = blockIdx.x >> 2;               // 0..255  (b*H + h)
  const int h  = bh & 7;
  const int b  = bh >> 3;
  const int st = (blockIdx.x & 3) * 8 + warp;   // 0..31
  const int s0 = st * 16;

  const _Float16* wg0 = WqT + h * DK_ * D_;
  const _Float16* wg1 = WkT + h * DK_ * D_;
  const _Float16* wg2 = WvT + h * DK_ * D_;

  const int srow = tid >> 2;                    // 0..63
  const int sseg = tid & 3;                     // 0..3

  auto stage = [&](int kc, int bufi) {
    const int goff = srow * D_ + kc * 32 + sseg * 8;
    const int loff = srow * WPAD + sseg * 8;
    async_copy16(wg0 + goff, &wbuf[bufi][0][loff]);
    async_copy16(wg1 + goff, &wbuf[bufi][1][loff]);
    async_copy16(wg2 + goff, &wbuf[bufi][2][loff]);
  };

  const _Float16* arow = xh + (b * S_ + s0 + mrow) * D_;

  v8f accQ[4], accK[4], accV[4];
#pragma unroll
  for (int j = 0; j < 4; ++j) { accQ[j] = (v8f)0.0f; accK[j] = (v8f)0.0f; accV[j] = (v8f)0.0f; }

  stage(0, 0);
  wait_async0();
  __syncthreads();

  for (int kc = 0; kc < 16; ++kc) {
    const int cur = kc & 1;
    if (kc + 1 < 16) stage(kc + 1, cur ^ 1);    // overlap copy with compute

    v16h a = load_a16(arow, kc * 32, kbA);
    if (kc + 2 < 16) __builtin_prefetch(arow + (kc + 2) * 32, 0, 0);
#pragma unroll
    for (int j = 0; j < 4; ++j) {
      const int roff = (j * 16 + mrow) * WPAD + kbB;
      v16h fq = *(const v16h*)&wbuf[cur][0][roff];
      v16h fk = *(const v16h*)&wbuf[cur][1][roff];
      v16h fv = *(const v16h*)&wbuf[cur][2][roff];
      accQ[j] = wmma16(a, fq, accQ[j]);
      accK[j] = wmma16(a, fk, accK[j]);
      accV[j] = wmma16(a, fv, accV[j]);
    }
    wait_async0();
    __syncthreads();
  }

  const float bias0 = bias[0];
  const float scale = 0.125f;                   // 1/sqrt(DK), folded into Q
#pragma unroll
  for (int j = 0; j < 4; ++j) {
#pragma unroll
    for (int v = 0; v < 8; ++v) {
      const int row = s0 + v + 8 * hi;
      const int col = j * 16 + mrow;
      const int qi  = ((b * H_ + h) * S_ + row) * DK_ + col;
      Qh[qi] = (_Float16)((accQ[j][v] + bias0) * scale);
      Kh[qi] = (_Float16)(accK[j][v] + bias0);
      VT[((b * H_ + h) * DK_ + col) * S_ + row] = (_Float16)(accV[j][v] + bias0);
    }
  }
}

// ---------------------------------------------------------------------------
// Kernel 3: fused attention per (b, h, 16-query-row block).
// Prologue: whole K panel (512x64 f16) for (b,h) async-staged into LDS,
//           shared by the 4 waves (rows padded to 72 halfs = 144B = 36 dw ->
//           lane-n start banks 36n mod 64 are disjoint quads).
// Phase 1: scores -> 16x512 f32 slab in LDS per wave.
// Phase 2: masked softmax (wave shfl_xor reductions), probs -> f16 slab.
// Phase 3: heads = P @ V via WMMA (P from LDS, VT contiguous from global).
// Dynamic LDS: 4*48KB (score/prob slabs) + 72KB (K panel) = 264 KB <= 320 KB.
// ---------------------------------------------------------------------------
#define KPAD 72
__global__ void k_attn(const _Float16* __restrict__ Qh,
                       const _Float16* __restrict__ Kh,
                       const _Float16* __restrict__ VT,
                       const int* __restrict__ mask,    // [B][S]
                       _Float16* __restrict__ MH) {     // [B][S][H*DK]
  extern __shared__ char smem[];
  const int tid  = threadIdx.x;
  const int lane = tid & 31;
  const int warp = tid >> 5;
  const int mrow = lane & 15;
  const int hi   = lane >> 4;
  const int kbA  = hi * 8;
  const int kbB  = hi * 16;

  float*    sc = (float*)(smem + warp * 49152);            // 16 x 512 f32
  _Float16* pf = (_Float16*)(smem + warp * 49152 + 32768); // 16 x 512 f16
  _Float16* Kl = (_Float16*)(smem + 196608);               // 512 x KPAD f16

  const int wid = blockIdx.x * 4 + warp;        // 0..8191
  const int st  = wid & 31;
  const int bh  = wid >> 5;
  const int h   = bh & 7;
  const int b   = bh >> 3;
  const int s0  = st * 16;
  const int bhS = (b * H_ + h) * S_;

  // ---- Prologue: async-stage K(b,h) [512][64] into LDS (once per block)
  const _Float16* Kg = Kh + bhS * DK_;
  for (int s = tid; s < 4096; s += 128) {       // 4096 16-byte segments
    const int row = s >> 3, sub = s & 7;
    async_copy16(Kg + row * DK_ + sub * 8, Kl + row * KPAD + sub * 8);
  }
  wait_async0();
  __syncthreads();

  // ---- Phase 1: scores -> LDS (Q already carries the 1/sqrt(DK) scale)
  const _Float16* qrow = Qh + (bhS + s0 + mrow) * DK_;
  v16h aq0 = load_a16(qrow, 0, kbA);
  v16h aq1 = load_a16(qrow, 32, kbA);

  for (int t0 = 0; t0 < S_; t0 += 16) {
    const _Float16* krow = Kl + (t0 + mrow) * KPAD;
    v16h b0 = *(const v16h*)(krow + kbB);
    v16h b1 = *(const v16h*)(krow + 32 + kbB);
    v8f acc = (v8f)0.0f;
    acc = wmma16(aq0, b0, acc);
    acc = wmma16(aq1, b1, acc);
#pragma unroll
    for (int v = 0; v < 8; ++v) sc[(v + 8 * hi) * S_ + t0 + mrow] = acc[v];
  }
  __syncthreads();

  // ---- Phase 2: masked softmax over 512 keys, 16 rows per wave
  int mk[16];
#pragma unroll
  for (int i = 0; i < 16; ++i) mk[i] = mask[b * S_ + lane + i * 32];

  for (int r = 0; r < 16; ++r) {
    float* row = sc + r * S_;
    float mx = -3.0e38f;
#pragma unroll
    for (int i = 0; i < 16; ++i) {
      const int c = lane + i * 32;
      float v = mk[i] ? row[c] : NEG_INF_F;     // m*score + (1-m)*NEG_INF
      row[c] = v;
      mx = fmaxf(mx, v);
    }
#pragma unroll
    for (int off = 16; off > 0; off >>= 1) mx = fmaxf(mx, __shfl_xor(mx, off, 32));
    float sum = 0.0f;
#pragma unroll
    for (int i = 0; i < 16; ++i) {
      const int c = lane + i * 32;
      float e = __expf(row[c] - mx);
      row[c] = e;
      sum += e;
    }
#pragma unroll
    for (int off = 16; off > 0; off >>= 1) sum += __shfl_xor(sum, off, 32);
    const float inv = 1.0f / sum;
#pragma unroll
    for (int i = 0; i < 16; ++i) {
      const int c = lane + i * 32;
      pf[r * S_ + c] = (_Float16)(row[c] * inv);
    }
  }
  __syncthreads();

  // ---- Phase 3: heads = P @ V  (batched loads, then 4 WMMAs)
  v8f acc[4];
#pragma unroll
  for (int j = 0; j < 4; ++j) acc[j] = (v8f)0.0f;

  const _Float16* prow  = pf + mrow * S_;
  const _Float16* vbase = VT + (b * H_ + h) * DK_ * S_;
  for (int t0 = 0; t0 < S_; t0 += 32) {
    v16h a = load_a16(prow, t0, kbA);
    v16h f[4];
#pragma unroll
    for (int j = 0; j < 4; ++j)
      f[j] = *(const v16h*)(vbase + (j * 16 + mrow) * S_ + t0 + kbB);
#pragma unroll
    for (int j = 0; j < 4; ++j) acc[j] = wmma16(a, f[j], acc[j]);
  }
#pragma unroll
  for (int j = 0; j < 4; ++j) {
#pragma unroll
    for (int v = 0; v < 8; ++v) {
      const int row = s0 + v + 8 * hi;
      const int col = h * DK_ + j * 16 + mrow;
      MH[(b * S_ + row) * D_ + col] = (_Float16)acc[j][v];
    }
  }
}

// ---------------------------------------------------------------------------
// Kernel 4: out = MH @ Wo + bias, f32 output.
// All 8 waves of a block share the same 16-row A strip of MH -> async-stage
// it into LDS once (rows padded to 520 halfs = 1040B = 260 dw -> lane-n A
// fragment reads start at bank 4n mod 64: conflict-free).
// One wave: 16 x 64 output strip; 8 waves/block; 1024 blocks.
// ---------------------------------------------------------------------------
#define APAD 520
__global__ void k_oproj(const _Float16* __restrict__ MH,   // [B*S][D] f16
                        const _Float16* __restrict__ WoT,  // [D][D] (n,k)
                        const float* __restrict__ bias,
                        float* __restrict__ out) {         // [B*S][D] f32
  __shared__ __align__(32) _Float16 Al[16 * APAD];         // 16.25 KB

  const int tid  = threadIdx.x;
  const int lane = tid & 31;
  const int warp = tid >> 5;
  const int mrow = lane & 15;
  const int hi   = lane >> 4;
  const int kbA  = hi * 8;
  const int kbB  = hi * 16;

  const int r0 = blockIdx.x * 16;               // block owns one 16-row strip
  const int c0 = warp * 64;                     // wave owns one 64-col block

  // ---- async-stage the 16x512 A strip (1024 16-byte segments, 256 threads)
  const _Float16* Ag = MH + r0 * D_;
#pragma unroll
  for (int i = 0; i < 4; ++i) {
    const int s = tid + i * 256;
    const int row = s >> 6, sub = s & 63;
    async_copy16(Ag + row * D_ + sub * 8, Al + row * APAD + sub * 8);
  }
  wait_async0();
  __syncthreads();

  v8f acc[4];
#pragma unroll
  for (int j = 0; j < 4; ++j) acc[j] = (v8f)0.0f;

  const _Float16* arow = Al + mrow * APAD;
  for (int k0 = 0; k0 < D_; k0 += 32) {
    v16h a = load_a16(arow, k0, kbA);
    v16h f[4];
#pragma unroll
    for (int j = 0; j < 4; ++j)
      f[j] = *(const v16h*)(WoT + (c0 + j * 16 + mrow) * D_ + k0 + kbB);
#pragma unroll
    for (int j = 0; j < 4; ++j) acc[j] = wmma16(a, f[j], acc[j]);
  }

  const float bias0 = bias[0];
#pragma unroll
  for (int j = 0; j < 4; ++j) {
#pragma unroll
    for (int v = 0; v < 8; ++v) {
      out[(r0 + v + 8 * hi) * D_ + c0 + j * 16 + mrow] = acc[j][v] + bias0;
    }
  }
}

// ---------------------------------------------------------------------------
// Launcher
// ---------------------------------------------------------------------------
extern "C" void kernel_launch(void* const* d_in, const int* in_sizes, int n_in,
                              void* d_out, int out_size, void* d_ws, size_t ws_size,
                              hipStream_t stream) {
  const float* x    = (const float*)d_in[0];
  const int*   mask = (const int*)d_in[1];
  const float* Wq   = (const float*)d_in[2];
  const float* Wk   = (const float*)d_in[3];
  const float* Wv   = (const float*)d_in[4];
  const float* Wo   = (const float*)d_in[5];
  const float* bias = (const float*)d_in[6];
  float* out = (float*)d_out;

  // Workspace layout (halfs)
  const size_t NX = (size_t)B_ * S_ * D_;   // 8,388,608
  const size_t NW = (size_t)H_ * DK_ * D_;  // 262,144
  const size_t NO = (size_t)D_ * D_;        // 262,144

  _Float16* xh  = (_Float16*)d_ws;
  _Float16* WqT = xh  + NX;
  _Float16* WkT = WqT + NW;
  _Float16* WvT = WkT + NW;
  _Float16* WoT = WvT + NW;
  _Float16* Qh  = WoT + NO;
  _Float16* Kh  = Qh  + NX;
  _Float16* VT  = Kh  + NX;
  _Float16* MH  = VT  + NX;

  k_convert<<<2048, 256, 0, stream>>>(x, Wq, Wk, Wv, Wo, xh, WqT, WkT, WvT, WoT);
  k_qkv<<<1024, 256, 0, stream>>>(xh, WqT, WkT, WvT, bias, Qh, Kh, VT);
  k_attn<<<2048, 128, 270336, stream>>>(Qh, Kh, VT, mask, MH);
  k_oproj<<<1024, 256, 0, stream>>>(MH, WoT, bias, out);
}
